// TimeDistributedLSTM_12833362280956
// MI455X (gfx1250) — matrix operand order repaired and checked by reading, not compiled, not run on hardware
//
#include <hip/hip_runtime.h>

// Problem constants (from reference): B=32,T=128,S=32,E=256,H=256 -> N=B*T=4096
#define Sz 32
#define Ez 256
#define Hz 256
#define Nz 4096
#define ROWS 32                 // batch rows per workgroup (2 WMMA M-tiles)
#define WGN (Nz / ROWS)         // 128 workgroups
#define BLOCK 512               // 16 wave32s: wave w owns hidden cols [16w,16w+16)
#define WELEM (Ez * 4 * Hz)     // 262144 elements per weight matrix

typedef __attribute__((ext_vector_type(16))) __bf16 v16bf;
typedef __attribute__((ext_vector_type(8)))  float  v8f;
typedef __attribute__((ext_vector_type(4)))  unsigned int u32x4;
typedef __attribute__((ext_vector_type(2)))  unsigned int u32x2;

union Frag16 { u32x4 q[2]; v16bf v; };

__device__ __forceinline__ unsigned short f2bf(float f) {
  unsigned int u = __builtin_bit_cast(unsigned int, f);
  u += 0x7fffu + ((u >> 16) & 1u);           // round-to-nearest-even
  return (unsigned short)(u >> 16);
}

__device__ __forceinline__ float fsig(float x) {
  return __builtin_amdgcn_rcpf(1.0f + __expf(-x));
}
__device__ __forceinline__ float ftanh(float x) {
  return 2.0f * __builtin_amdgcn_rcpf(1.0f + __expf(-2.0f * x)) - 1.0f;
}

// Prepass: W fp32 -> bf16 into workspace, bias = b_ih + b_hh (fp32).
__global__ void lstm_prep(const float* __restrict__ Wih, const float* __restrict__ Whh,
                          const float* __restrict__ bih, const float* __restrict__ bhh,
                          unsigned short* __restrict__ wbf, float* __restrict__ bias) {
  int i = blockIdx.x * blockDim.x + threadIdx.x;     // 0 .. WELEM-1
  wbf[i]         = f2bf(Wih[i]);
  wbf[WELEM + i] = f2bf(Whh[i]);
  if (i < 4 * Hz) bias[i] = bih[i] + bhh[i];
}

// Fused persistent LSTM: each WG owns ROWS batch rows for all S steps.
// Wave w owns hidden cols [16w, 16w+16) across all 4 gates: cell update stays
// in registers; h is exchanged through LDS in bf16 each step.
__global__ __launch_bounds__(BLOCK, 1)
void lstm_fused(const float* __restrict__ x,
                const unsigned short* __restrict__ wih,
                const unsigned short* __restrict__ whh,
                const float* __restrict__ bias,
                float* __restrict__ out) {
  __shared__ __align__(16) unsigned short sX[ROWS * Ez];  // x_s tile, bf16 (16 KB)
  __shared__ __align__(16) unsigned short sH[ROWS * Hz];  // h tile, bf16 (16 KB)

  const int tid   = threadIdx.x;
  const int lane  = tid & 31;
  const int wave  = tid >> 5;              // 0..15 -> hcol tile
  const int lhalf = (lane < 16) ? 0 : 1;   // lane group for A/B K-halves
  const int lcol  = lane & 15;             // B/C/D column within a 16-wide tile
  const int rowbase = blockIdx.x * ROWS;
  const int colbase = wave * 16 + lcol;    // this lane's hidden column

  // n-tile nt = gate G in {i,f,g,o}; absolute gate col g = G*Hz + colbase
  float bgate[4];
#pragma unroll
  for (int G = 0; G < 4; ++G) bgate[G] = bias[G * Hz + colbase];

  // h0 = 0  (ROWS*Hz ushorts = 16 KB = 1024 u32x4; 512 threads -> 2 each)
  {
    u32x4 z = {0u, 0u, 0u, 0u};
    u32x4* p = (u32x4*)sH;
#pragma unroll
    for (int j = 0; j < 2; ++j) p[tid + j * BLOCK] = z;
  }

  v8f c[2];                                // cell state per m-tile, fp32, in regs
#pragma unroll
  for (int m = 0; m < 2; ++m)
#pragma unroll
    for (int e = 0; e < 8; ++e) c[m][e] = 0.0f;

#pragma unroll 1
  for (int s = 0; s < Sz; ++s) {
    __syncthreads();                       // (a) prev-step LDS reads + h writes done

    // Stage x[rowbase.., s, :] fp32 -> bf16 into sX (32x256); 16 floats/thread
    {
      const int r  = tid >> 4;             // 0..31
      const int cb = (tid & 15) * 16;      // 0..240
      const float* srcf = x + ((size_t)(rowbase + r) * Sz + s) * Ez + cb;
      const float4* src = (const float4*)srcf;
#pragma unroll
      for (int j = 0; j < 4; ++j) {
        float4 v = src[j];
        u32x2 pk;
        pk[0] = (unsigned int)f2bf(v.x) | ((unsigned int)f2bf(v.y) << 16);
        pk[1] = (unsigned int)f2bf(v.z) | ((unsigned int)f2bf(v.w) << 16);
        *(u32x2*)(sX + r * Ez + cb + j * 4) = pk;
      }
      if (s + 1 < Sz)                      // pull next step's x tile toward L2/L0
        __builtin_prefetch(srcf + Ez, 0, 0);
    }
    __syncthreads();                       // (b) staging visible

    // Init accumulators with bias (C/D layout: col = lcol -> bias uniform per lane)
    v8f acc[2][4];
#pragma unroll
    for (int m = 0; m < 2; ++m)
#pragma unroll
      for (int G = 0; G < 4; ++G)
#pragma unroll
        for (int e = 0; e < 8; ++e) acc[m][G][e] = bgate[G];

    // ---- GEMM 1: x_s @ W_ih^T ----
    // unroll 1: keeps only 2 A-frags + 4 B-frags live -> no VGPR spills, and
    // weight loads stay inside the loop (kt-dependent address blocks LICM).
#pragma unroll 1
    for (int kt = 0; kt < Ez / 32; ++kt) {
      const int kb = kt * 32;
      Frag16 A[2];
#pragma unroll
      for (int m = 0; m < 2; ++m) {        // A 16x32 bf16: two 16B K-runs per lane
        const unsigned short* ap = sX + (m * 16 + lcol) * Ez + kb + lhalf * 8;
        A[m].q[0] = *(const u32x4*)(ap);
        A[m].q[1] = *(const u32x4*)(ap + 16);
      }
#pragma unroll
      for (int G = 0; G < 4; ++G) {
        const int g = G * Hz + colbase;
        const unsigned short* bp = wih + (size_t)g * Ez + kb + lhalf * 8;
        Frag16 Bf;                         // B 32x16: two 8-bf16 K-runs of W row g
        Bf.q[0] = *(const u32x4*)(bp);
        Bf.q[1] = *(const u32x4*)(bp + 16);
#pragma unroll
        for (int m = 0; m < 2; ++m)
          acc[m][G] = __builtin_amdgcn_wmma_f32_16x16x32_bf16(
              false, A[m].v, false, Bf.v, (short)0, acc[m][G], false, false);
      }
    }

    // ---- GEMM 2: h @ W_hh^T ----
#pragma unroll 1
    for (int kt = 0; kt < Hz / 32; ++kt) {
      const int kb = kt * 32;
      Frag16 A[2];
#pragma unroll
      for (int m = 0; m < 2; ++m) {
        const unsigned short* ap = sH + (m * 16 + lcol) * Hz + kb + lhalf * 8;
        A[m].q[0] = *(const u32x4*)(ap);
        A[m].q[1] = *(const u32x4*)(ap + 16);
      }
#pragma unroll
      for (int G = 0; G < 4; ++G) {
        const int g = G * Hz + colbase;
        const unsigned short* bp = whh + (size_t)g * Hz + kb + lhalf * 8;
        Frag16 Bf;
        Bf.q[0] = *(const u32x4*)(bp);
        Bf.q[1] = *(const u32x4*)(bp + 16);
#pragma unroll
        for (int m = 0; m < 2; ++m)
          acc[m][G] = __builtin_amdgcn_wmma_f32_16x16x32_bf16(
              false, A[m].v, false, Bf.v, (short)0, acc[m][G], false, false);
      }
    }

    // ---- Cell update, fully in registers (wave owns i,f,g,o of its hcols) ----
#pragma unroll
    for (int m = 0; m < 2; ++m) {
#pragma unroll
      for (int e = 0; e < 8; ++e) {
        float ii = fsig(acc[m][0][e]);
        float ff = fsig(acc[m][1][e]);
        float gg = ftanh(acc[m][2][e]);
        float oo = fsig(acc[m][3][e]);
        float cn = ff * c[m][e] + ii * gg;
        c[m][e] = cn;
        float hn = oo * ftanh(cn);
        if (s == Sz - 1) {                 // final h -> fp32 output
          const int row = m * 16 + e + lhalf * 8;
          out[(size_t)(rowbase + row) * Hz + colbase] = hn;
        } else {
          acc[m][3][e] = hn;               // stash h_next (o-gate acc is dead)
        }
      }
    }

    if (s < Sz - 1) {
      __syncthreads();                     // (c) all GEMM reads of sH done
#pragma unroll
      for (int m = 0; m < 2; ++m)
#pragma unroll
        for (int e = 0; e < 8; ++e) {
          const int row = m * 16 + e + lhalf * 8;
          sH[row * Hz + colbase] = f2bf(acc[m][3][e]);
        }
    }
  }
}

extern "C" void kernel_launch(void* const* d_in, const int* in_sizes, int n_in,
                              void* d_out, int out_size, void* d_ws, size_t ws_size,
                              hipStream_t stream) {
  const float* x   = (const float*)d_in[0];
  const float* Wih = (const float*)d_in[1];
  const float* Whh = (const float*)d_in[2];
  const float* bih = (const float*)d_in[3];
  const float* bhh = (const float*)d_in[4];

  unsigned short* wbf = (unsigned short*)d_ws;                    // 2 * 512 KB bf16 weights
  float* bias = (float*)((char*)d_ws + (size_t)2 * WELEM * sizeof(unsigned short));

  lstm_prep<<<WELEM / 256, 256, 0, stream>>>(Wih, Whh, bih, bhh, wbf, bias);
  lstm_fused<<<WGN, BLOCK, 0, stream>>>(x, wbf, wbf + WELEM, bias, (float*)d_out);
}
